// TopKSelection_66408784330770
// MI455X (gfx1250) — compile-verified, matrix operand order checked by reading, not run on hardware
//
#include <hip/hip_runtime.h>
#include <cstdint>
#include <cstddef>

// ---------------------------------------------------------------------------
// Types for CDNA5 WMMA (wave32): v_wmma_f32_16x16x32_bf16
// ---------------------------------------------------------------------------
typedef __attribute__((ext_vector_type(16))) __bf16          v16bf;
typedef __attribute__((ext_vector_type(8)))  float           v8f;
typedef __attribute__((ext_vector_type(8)))  unsigned short  u16x8;
typedef __attribute__((ext_vector_type(4)))  unsigned short  u16x4;

union Frag {
    u16x8 h[2];   // two 16B halves: K-runs {8g..8g+7} and {16+8g..16+8g+7}
    v16bf v;
};

__device__ __forceinline__ unsigned short f2bf(float f) {
    unsigned int u = __float_as_uint(f);
    u += 0x7FFFu + ((u >> 16) & 1u);           // round-to-nearest-even
    return (unsigned short)(u >> 16);
}
__device__ __forceinline__ float bf2f(unsigned short h) {
    return __uint_as_float(((unsigned int)h) << 16);
}
__device__ __forceinline__ void split2(float f, unsigned short& hi, unsigned short& lo) {
    const unsigned short h = f2bf(f);
    hi = h;
    lo = f2bf(f - bf2f(h));
}

#define WMMA_BF16(A, B, C) \
    __builtin_amdgcn_wmma_f32_16x16x32_bf16(false, (A), false, (B), (short)0, (C), false, false)

// gfx1250 async global->LDS staging (ASYNCcnt-tracked), with safe fallback.
#if defined(__HIP_DEVICE_COMPILE__) && __has_builtin(__builtin_amdgcn_global_load_async_to_lds_b128)
  #define ASYNC_STAGE 1
#else
  #define ASYNC_STAGE 0
#endif

__device__ __forceinline__ void wait_async_loads() {
#if ASYNC_STAGE
  #if __has_builtin(__builtin_amdgcn_s_wait_asynccnt)
    __builtin_amdgcn_s_wait_asynccnt(0);
  #else
    asm volatile("s_wait_asynccnt 0x0" ::: "memory");
  #endif
#endif
}

__device__ __forceinline__ void stage16(const unsigned short* __restrict__ g,
                                        unsigned short* l) {
#if ASYNC_STAGE
    // Builtin signature (from hipcc diagnostic): param 0 is
    // 'int __vector(4) __device__ *' (GCC vector_size(16), target AS 1).
    typedef int v4i_vs __attribute__((vector_size(16)));
    __builtin_amdgcn_global_load_async_to_lds_b128(
        (__attribute__((address_space(1))) v4i_vs*)g,
        (__attribute__((address_space(3))) v4i_vs*)l,
        0, 0);
#else
    *reinterpret_cast<u16x8*>(l) = *reinterpret_cast<const u16x8*>(g);
#endif
}

// ---------------------------------------------------------------------------
// Kernel 1: importance[b,s] = x[b,s,:] . w_score + b_score    (1 wave / row)
// ---------------------------------------------------------------------------
__global__ __launch_bounds__(256) void importance_kernel(
    const float* __restrict__ x, const float* __restrict__ w,
    const float* __restrict__ bsc, float* __restrict__ imp, int D)
{
    const int row  = blockIdx.x * 8 + (threadIdx.x >> 5);
    const int lane = threadIdx.x & 31;
    const float4* xr = reinterpret_cast<const float4*>(&x[(size_t)row * D]);
    const float4* wr = reinterpret_cast<const float4*>(w);
    float s = 0.f;
    for (int i = lane; i < D / 4; i += 32) {
        const float4 a = xr[i], b = wr[i];
        s += a.x * b.x + a.y * b.y + a.z * b.z + a.w * b.w;
    }
    #pragma unroll
    for (int off = 16; off > 0; off >>= 1) s += __shfl_xor(s, off, 32);
    if (lane == 0) imp[row] = s + bsc[0];
}

// ---------------------------------------------------------------------------
// Kernel 2: exact jax.lax.top_k ordering via rank = #{j : v[j]>v[i] or
// (v[j]==v[i] and j<i)}.  One block per batch, values cached in LDS.
// ---------------------------------------------------------------------------
__global__ __launch_bounds__(256) void topk_kernel(
    const float* __restrict__ imp, int* __restrict__ topidx,
    int* __restrict__ out_idx, int S, int k)
{
    __shared__ float simp[4096];
    const int b = blockIdx.x, tid = threadIdx.x;
    for (int i = tid; i < S; i += 256) simp[i] = imp[b * S + i];
    __syncthreads();
    for (int i = tid; i < S; i += 256) {
        const float v = simp[i];
        int cnt = 0;
        for (int j = 0; j < S; ++j) {
            const float u = simp[j];
            cnt += (int)((u > v) | ((u == v) & (j < i)));
        }
        if (cnt < k) {
            topidx[b * k + cnt]  = i;
            out_idx[b * k + cnt] = i;
        }
    }
}

// ---------------------------------------------------------------------------
// Kernel 3: one-time fp32 -> bf16 hi/lo plane split (row-major, no transpose)
// ---------------------------------------------------------------------------
__global__ __launch_bounds__(256) void split_planes_kernel(
    const float* __restrict__ X, unsigned short* __restrict__ hiP,
    unsigned short* __restrict__ loP)
{
    const size_t i = (size_t)blockIdx.x * 256 + threadIdx.x;   // float4 index
    const float4 f = reinterpret_cast<const float4*>(X)[i];
    unsigned short h0, h1, h2, h3, l0, l1, l2, l3;
    split2(f.x, h0, l0);
    split2(f.y, h1, l1);
    split2(f.z, h2, l2);
    split2(f.w, h3, l3);
    reinterpret_cast<u16x4*>(hiP)[i] = (u16x4){h0, h1, h2, h3};
    reinterpret_cast<u16x4*>(loP)[i] = (u16x4){l0, l1, l2, l3};
}

// ---------------------------------------------------------------------------
// Kernel 4: one-time weight split+transpose: W[K][N] fp32 -> hiT/loT [N][K]
// 32x32 LDS tile so both global accesses are coalesced.
// ---------------------------------------------------------------------------
__global__ __launch_bounds__(256) void split_transpose_kernel(
    const float* __restrict__ W, unsigned short* __restrict__ hiT,
    unsigned short* __restrict__ loT, int K, int N)
{
    __shared__ unsigned short th[32][33], tl[32][33];
    const int n0 = blockIdx.x * 32;
    const int k0 = blockIdx.y * 32;
    const int tid = threadIdx.x;
    #pragma unroll
    for (int it = 0; it < 4; ++it) {
        const int lin = tid + it * 256;
        const int r = lin >> 5;          // k within tile
        const int c = lin & 31;          // n within tile
        split2(W[(size_t)(k0 + r) * N + n0 + c], th[r][c], tl[r][c]);
    }
    __syncthreads();
    #pragma unroll
    for (int it = 0; it < 4; ++it) {
        const int lin = tid + it * 256;
        const int nr = lin >> 5;         // n within tile
        const int kc = lin & 31;         // k within tile
        hiT[(size_t)(n0 + nr) * K + k0 + kc] = th[kc][nr];
        loT[(size_t)(n0 + nr) * K + k0 + kc] = tl[kc][nr];
    }
}

// ---------------------------------------------------------------------------
// Kernel 5: gather selected rows directly in plane form
// ---------------------------------------------------------------------------
__global__ __launch_bounds__(256) void gather_planes_kernel(
    const unsigned short* __restrict__ xHi, const unsigned short* __restrict__ xLo,
    const int* __restrict__ topidx, unsigned short* __restrict__ selHi,
    unsigned short* __restrict__ selLo, int S, int D)
{
    const int row = blockIdx.x;          // b*64 + i
    const int b   = row >> 6;
    const int src = topidx[row];
    const u16x8* sH = reinterpret_cast<const u16x8*>(&xHi[((size_t)b * S + src) * D]);
    const u16x8* sL = reinterpret_cast<const u16x8*>(&xLo[((size_t)b * S + src) * D]);
    u16x8* dH = reinterpret_cast<u16x8*>(&selHi[(size_t)row * D]);
    u16x8* dL = reinterpret_cast<u16x8*>(&selLo[(size_t)row * D]);
    const int t = threadIdx.x;           // D/8 = 128 chunks per plane
    if (t < 128)      dH[t] = sH[t];
    else              dL[t - 128] = sL[t - 128];
}

// ---------------------------------------------------------------------------
// Kernel 6: Y = X@W + bias from pre-split bf16 planes, bf16x3 WMMA core.
// A: hi/lo planes [M][K]; B: hi/lo transposed planes [N][K].
// Block tile 128x128, K-step 32, double-buffered LDS (64KB), 8 waves
// (4M x 2N), wave tile 32x64.  Staging uses gfx1250 async global->LDS
// when available.  Epilogue writes fp32 (Yf32) or bf16 hi/lo planes.
// M,N multiples of 128; K multiple of 32.
// ---------------------------------------------------------------------------
#define GBM 128
#define GBN 128
#define GBK 32

__global__ __launch_bounds__(256) void gemm_planes_kernel(
    const unsigned short* __restrict__ Ahi, const unsigned short* __restrict__ Alo,
    const unsigned short* __restrict__ BhiT, const unsigned short* __restrict__ BloT,
    const float* __restrict__ bias, float* __restrict__ Yf32,
    unsigned short* __restrict__ Yhi, unsigned short* __restrict__ Ylo,
    int M, int N, int K)
{
    __shared__ unsigned short sA[2][2][GBM][GBK];   // [buf][hi/lo][m][k]
    __shared__ unsigned short sB[2][2][GBN][GBK];   // [buf][hi/lo][n][k]

    const int tid  = threadIdx.x;
    const int lane = tid & 31;
    const int wave = tid >> 5;
    const int wm   = wave & 3;                 // 4 waves along M
    const int wn   = wave >> 2;                // 2 waves along N
    const int l16  = lane & 15;
    const int lh8  = (lane >> 4) * 8;          // 0 or 8

    const int m0 = blockIdx.x * GBM;
    const int n0 = blockIdx.y * GBN;

    auto stage = [&](int buf, int kb) {
        #pragma unroll
        for (int it = 0; it < 2; ++it) {
            const int lin = tid + it * 256;    // 0..511
            const int r  = lin >> 2;           // 0..127
            const int co = (lin & 3) << 3;     // 0,8,16,24
            stage16(&Ahi [(size_t)(m0 + r) * K + kb + co], &sA[buf][0][r][co]);
            stage16(&Alo [(size_t)(m0 + r) * K + kb + co], &sA[buf][1][r][co]);
            stage16(&BhiT[(size_t)(n0 + r) * K + kb + co], &sB[buf][0][r][co]);
            stage16(&BloT[(size_t)(n0 + r) * K + kb + co], &sB[buf][1][r][co]);
        }
    };

    v8f acc[2][4];
    #pragma unroll
    for (int i = 0; i < 2; ++i)
        #pragma unroll
        for (int j = 0; j < 4; ++j)
            acc[i][j] = (v8f){0.f,0.f,0.f,0.f,0.f,0.f,0.f,0.f};

    stage(0, 0);
    wait_async_loads();
    __syncthreads();

    for (int kb = 0; kb < K; kb += GBK) {
        const int buf = (kb >> 5) & 1;
        if (kb + GBK < K) stage(buf ^ 1, kb + GBK);   // prefetch next tile

        Frag aH[2], aL[2];
        #pragma unroll
        for (int i = 0; i < 2; ++i) {
            const int row = wm * 32 + i * 16 + l16;
            aH[i].h[0] = *reinterpret_cast<const u16x8*>(&sA[buf][0][row][lh8]);
            aH[i].h[1] = *reinterpret_cast<const u16x8*>(&sA[buf][0][row][16 + lh8]);
            aL[i].h[0] = *reinterpret_cast<const u16x8*>(&sA[buf][1][row][lh8]);
            aL[i].h[1] = *reinterpret_cast<const u16x8*>(&sA[buf][1][row][16 + lh8]);
        }
        #pragma unroll
        for (int j = 0; j < 4; ++j) {
            const int col = wn * 64 + j * 16 + l16;
            Frag bH, bL;
            bH.h[0] = *reinterpret_cast<const u16x8*>(&sB[buf][0][col][lh8]);
            bH.h[1] = *reinterpret_cast<const u16x8*>(&sB[buf][0][col][16 + lh8]);
            bL.h[0] = *reinterpret_cast<const u16x8*>(&sB[buf][1][col][lh8]);
            bL.h[1] = *reinterpret_cast<const u16x8*>(&sB[buf][1][col][16 + lh8]);
            #pragma unroll
            for (int i = 0; i < 2; ++i) {
                acc[i][j] = WMMA_BF16(aH[i].v, bH.v, acc[i][j]);
                acc[i][j] = WMMA_BF16(aH[i].v, bL.v, acc[i][j]);
                acc[i][j] = WMMA_BF16(aL[i].v, bH.v, acc[i][j]);
            }
        }
        wait_async_loads();
        __syncthreads();
    }

    if (Yf32 != nullptr) {
        #pragma unroll
        for (int i = 0; i < 2; ++i) {
            #pragma unroll
            for (int j = 0; j < 4; ++j) {
                const int n  = n0 + wn * 64 + j * 16 + l16;
                const float bv = bias[n];
                #pragma unroll
                for (int r = 0; r < 8; ++r) {
                    const int m = m0 + wm * 32 + i * 16 + lh8 + r;
                    Yf32[(size_t)m * N + n] = acc[i][j][r] + bv;
                }
            }
        }
    } else {
        #pragma unroll
        for (int i = 0; i < 2; ++i) {
            #pragma unroll
            for (int j = 0; j < 4; ++j) {
                const int n  = n0 + wn * 64 + j * 16 + l16;
                const float bv = bias[n];
                #pragma unroll
                for (int r = 0; r < 8; ++r) {
                    const int m = m0 + wm * 32 + i * 16 + lh8 + r;
                    unsigned short hi, lo;
                    split2(acc[i][j][r] + bv, hi, lo);
                    Yhi[(size_t)m * N + n] = hi;
                    Ylo[(size_t)m * N + n] = lo;
                }
            }
        }
    }
}

// ---------------------------------------------------------------------------
// Kernel 7: attention.  One wave per (b, h, 16-query tile).
// q/k/v arrive as pre-split bf16 hi/lo planes (no conversion VALU here):
// staging is pure u16x8 copies plus a ushort transpose for V^T.
// scores via bf16x3 WMMA; mask by original key position; exact fp32
// softmax; out = P@V with P bf16, V hi+lo.  Output written as hi/lo
// planes for the O-projection GEMM.
// ---------------------------------------------------------------------------
__global__ __launch_bounds__(32) void sel_attn_kernel(
    const unsigned short* __restrict__ qHiP, const unsigned short* __restrict__ qLoP,
    const unsigned short* __restrict__ kHiP, const unsigned short* __restrict__ kLoP,
    const unsigned short* __restrict__ vHiP, const unsigned short* __restrict__ vLoP,
    const int* __restrict__ topidx,
    unsigned short* __restrict__ oHi, unsigned short* __restrict__ oLo, int S)
{
    __shared__ unsigned short qHi[16][64], qLo[16][64];
    __shared__ unsigned short kHi[64][64], kLo[64][64];   // [key][dim]
    __shared__ unsigned short vHiT[64][72], vLoT[64][72]; // [dim][key], padded
    __shared__ unsigned short pB[16][64];                 // probs bf16 [m][key]
    __shared__ int kpos[64];

    const int lane = threadIdx.x;
    const int l16  = lane & 15;
    const int lh8  = (lane >> 4) * 8;
    const int s0 = blockIdx.x * 16;
    const int h  = blockIdx.y;
    const int b  = blockIdx.z;

    for (int i = lane; i < 64; i += 32) kpos[i] = topidx[b * 64 + i];

    // q tile: 16 rows x 8 chunks of u16x8 per plane
    for (int c = lane; c < 128; c += 32) {
        const int m = c >> 3, ch = (c & 7) * 8;
        const size_t g = ((size_t)(b * S + s0 + m)) * 1024 + h * 64 + ch;
        *reinterpret_cast<u16x8*>(&qHi[m][ch]) = *reinterpret_cast<const u16x8*>(&qHiP[g]);
        *reinterpret_cast<u16x8*>(&qLo[m][ch]) = *reinterpret_cast<const u16x8*>(&qLoP[g]);
    }
    // k tile: 64 rows x 8 chunks per plane
    for (int c = lane; c < 512; c += 32) {
        const int kk = c >> 3, ch = (c & 7) * 8;
        const size_t g = ((size_t)(b * 64 + kk)) * 1024 + h * 64 + ch;
        *reinterpret_cast<u16x8*>(&kHi[kk][ch]) = *reinterpret_cast<const u16x8*>(&kHiP[g]);
        *reinterpret_cast<u16x8*>(&kLo[kk][ch]) = *reinterpret_cast<const u16x8*>(&kLoP[g]);
    }
    // v tile transposed to [dim][key] (coalesced global reads)
    for (int i = lane; i < 4096; i += 32) {
        const int kk = i >> 6, d = i & 63;
        const size_t g = ((size_t)(b * 64 + kk)) * 1024 + h * 64 + d;
        vHiT[d][kk] = vHiP[g];
        vLoT[d][kk] = vLoP[g];
    }
    __syncthreads();

    // --- scores: 4 key-tiles x 2 K-steps, bf16x3 ---
    v8f sc[4];
    #pragma unroll
    for (int j = 0; j < 4; ++j) sc[j] = (v8f){0.f,0.f,0.f,0.f,0.f,0.f,0.f,0.f};

    #pragma unroll
    for (int j = 0; j < 4; ++j) {
        #pragma unroll
        for (int ks = 0; ks < 2; ++ks) {
            const int ko = ks * 32;
            Frag ah, al, bh, bl;
            ah.h[0] = *reinterpret_cast<const u16x8*>(&qHi[l16][ko + lh8]);
            ah.h[1] = *reinterpret_cast<const u16x8*>(&qHi[l16][ko + 16 + lh8]);
            al.h[0] = *reinterpret_cast<const u16x8*>(&qLo[l16][ko + lh8]);
            al.h[1] = *reinterpret_cast<const u16x8*>(&qLo[l16][ko + 16 + lh8]);
            const int col = j * 16 + l16;
            bh.h[0] = *reinterpret_cast<const u16x8*>(&kHi[col][ko + lh8]);
            bh.h[1] = *reinterpret_cast<const u16x8*>(&kHi[col][ko + 16 + lh8]);
            bl.h[0] = *reinterpret_cast<const u16x8*>(&kLo[col][ko + lh8]);
            bl.h[1] = *reinterpret_cast<const u16x8*>(&kLo[col][ko + 16 + lh8]);
            sc[j] = WMMA_BF16(ah.v, bh.v, sc[j]);
            sc[j] = WMMA_BF16(ah.v, bl.v, sc[j]);
            sc[j] = WMMA_BF16(al.v, bh.v, sc[j]);
        }
    }

    // --- scale + causal mask against original positions ---
    #pragma unroll
    for (int j = 0; j < 4; ++j) {
        const int kp = kpos[j * 16 + l16];
        #pragma unroll
        for (int r = 0; r < 8; ++r) {
            const int qp = s0 + lh8 + r;
            const float v = sc[j][r] * 0.125f;      // 1/sqrt(64)
            sc[j][r] = (qp >= kp) ? v : -1.0e9f;
        }
    }

    // --- softmax per row (row m lives in 16 lanes of one half, vgpr m%8) ---
    #pragma unroll
    for (int r = 0; r < 8; ++r) {
        float mx = fmaxf(fmaxf(sc[0][r], sc[1][r]), fmaxf(sc[2][r], sc[3][r]));
        #pragma unroll
        for (int off = 1; off < 16; off <<= 1) mx = fmaxf(mx, __shfl_xor(mx, off, 32));
        float sum = 0.f;
        #pragma unroll
        for (int j = 0; j < 4; ++j) { const float e = __expf(sc[j][r] - mx); sc[j][r] = e; sum += e; }
        #pragma unroll
        for (int off = 1; off < 16; off <<= 1) sum += __shfl_xor(sum, off, 32);
        const float inv = 1.f / sum;
        #pragma unroll
        for (int j = 0; j < 4; ++j) sc[j][r] *= inv;
    }

    // --- restage P as bf16 A-operand ---
    #pragma unroll
    for (int j = 0; j < 4; ++j)
        #pragma unroll
        for (int r = 0; r < 8; ++r)
            pB[lh8 + r][j * 16 + l16] = f2bf(sc[j][r]);
    __syncthreads();

    // --- out = P @ V (P bf16, V hi+lo) ---
    v8f oc[4];
    #pragma unroll
    for (int j = 0; j < 4; ++j) oc[j] = (v8f){0.f,0.f,0.f,0.f,0.f,0.f,0.f,0.f};

    #pragma unroll
    for (int j2 = 0; j2 < 4; ++j2) {
        #pragma unroll
        for (int ks = 0; ks < 2; ++ks) {
            const int ko = ks * 32;
            Frag p, vh, vl;
            p.h[0] = *reinterpret_cast<const u16x8*>(&pB[l16][ko + lh8]);
            p.h[1] = *reinterpret_cast<const u16x8*>(&pB[l16][ko + 16 + lh8]);
            const int col = j2 * 16 + l16;
            vh.h[0] = *reinterpret_cast<const u16x8*>(&vHiT[col][ko + lh8]);
            vh.h[1] = *reinterpret_cast<const u16x8*>(&vHiT[col][ko + 16 + lh8]);
            vl.h[0] = *reinterpret_cast<const u16x8*>(&vLoT[col][ko + lh8]);
            vl.h[1] = *reinterpret_cast<const u16x8*>(&vLoT[col][ko + 16 + lh8]);
            oc[j2] = WMMA_BF16(p.v, vh.v, oc[j2]);
            oc[j2] = WMMA_BF16(p.v, vl.v, oc[j2]);
        }
    }

    #pragma unroll
    for (int j2 = 0; j2 < 4; ++j2) {
        #pragma unroll
        for (int r = 0; r < 8; ++r) {
            const int m = lh8 + r;
            const int d = j2 * 16 + l16;
            const size_t idx = ((size_t)(b * S + s0 + m)) * 1024 + h * 64 + d;
            unsigned short hi, lo;
            split2(oc[j2][r], hi, lo);
            oHi[idx] = hi;
            oLo[idx] = lo;
        }
    }
}

// ---------------------------------------------------------------------------
// Host launcher
// ---------------------------------------------------------------------------
extern "C" void kernel_launch(void* const* d_in, const int* /*in_sizes*/, int /*n_in*/,
                              void* d_out, int /*out_size*/, void* d_ws, size_t /*ws_size*/,
                              hipStream_t stream)
{
    const float* x       = (const float*)d_in[0];
    const float* w_score = (const float*)d_in[1];
    const float* b_score = (const float*)d_in[2];
    const float* w_q     = (const float*)d_in[3];
    const float* b_q     = (const float*)d_in[4];
    const float* w_k     = (const float*)d_in[5];
    const float* b_k     = (const float*)d_in[6];
    const float* w_v     = (const float*)d_in[7];
    const float* b_v     = (const float*)d_in[8];
    const float* w_o     = (const float*)d_in[9];
    const float* b_o     = (const float*)d_in[10];

    const int B = 4, S = 4096, D = 1024, H = 16, TK = 64;
    const int Mq = B * S;                 // 16384 rows

    char* ws = (char*)d_ws;
    size_t off = 0;
    auto take = [&](size_t bytes) {
        char* p = ws + off;
        off += (bytes + 255) & ~(size_t)255;
        return p;
    };
    float*          imp    = (float*)take((size_t)B * S * sizeof(float));
    int*            topidx = (int*)  take((size_t)B * TK * sizeof(int));
    unsigned short* xHi    = (unsigned short*)take((size_t)Mq * D * 2);
    unsigned short* xLo    = (unsigned short*)take((size_t)Mq * D * 2);
    unsigned short* selHi  = (unsigned short*)take((size_t)B * TK * D * 2);
    unsigned short* selLo  = (unsigned short*)take((size_t)B * TK * D * 2);
    unsigned short* wqHi   = (unsigned short*)take((size_t)D * D * 2);
    unsigned short* wqLo   = (unsigned short*)take((size_t)D * D * 2);
    unsigned short* wkHi   = (unsigned short*)take((size_t)D * D * 2);
    unsigned short* wkLo   = (unsigned short*)take((size_t)D * D * 2);
    unsigned short* wvHi   = (unsigned short*)take((size_t)D * D * 2);
    unsigned short* wvLo   = (unsigned short*)take((size_t)D * D * 2);
    unsigned short* woHi   = (unsigned short*)take((size_t)D * D * 2);
    unsigned short* woLo   = (unsigned short*)take((size_t)D * D * 2);
    unsigned short* qHiP   = (unsigned short*)take((size_t)Mq * D * 2);
    unsigned short* qLoP   = (unsigned short*)take((size_t)Mq * D * 2);
    unsigned short* kHiP   = (unsigned short*)take((size_t)B * TK * D * 2);
    unsigned short* kLoP   = (unsigned short*)take((size_t)B * TK * D * 2);
    unsigned short* vHiP   = (unsigned short*)take((size_t)B * TK * D * 2);
    unsigned short* vLoP   = (unsigned short*)take((size_t)B * TK * D * 2);
    unsigned short* oHi    = (unsigned short*)take((size_t)Mq * D * 2);
    unsigned short* oLo    = (unsigned short*)take((size_t)Mq * D * 2);

    float* out     = (float*)d_out;
    int*   out_idx = (int*)(out + (size_t)Mq * D);   // tuple tail: top_k_indices

    importance_kernel<<<Mq / 8, 256, 0, stream>>>(x, w_score, b_score, imp, D);
    topk_kernel<<<B, 256, 0, stream>>>(imp, topidx, out_idx, S, TK);

    // one-time operand pre-splitting (fp32 -> bf16 hi/lo planes)
    split_planes_kernel<<<(Mq * D) / (4 * 256), 256, 0, stream>>>(x, xHi, xLo);
    split_transpose_kernel<<<dim3(D / 32, D / 32), 256, 0, stream>>>(w_q, wqHi, wqLo, D, D);
    split_transpose_kernel<<<dim3(D / 32, D / 32), 256, 0, stream>>>(w_k, wkHi, wkLo, D, D);
    split_transpose_kernel<<<dim3(D / 32, D / 32), 256, 0, stream>>>(w_v, wvHi, wvLo, D, D);
    split_transpose_kernel<<<dim3(D / 32, D / 32), 256, 0, stream>>>(w_o, woHi, woLo, D, D);
    gather_planes_kernel<<<B * TK, 256, 0, stream>>>(xHi, xLo, topidx, selHi, selLo, S, D);

    // projections: async-staged, double-buffered, bf16x3 WMMA.
    // q/k/v write bf16 hi/lo planes (split fused into epilogue).
    gemm_planes_kernel<<<dim3(Mq / GBM, D / GBN), 256, 0, stream>>>(
        xHi, xLo, wqHi, wqLo, b_q, nullptr, qHiP, qLoP, Mq, D, D);
    gemm_planes_kernel<<<dim3((B * TK) / GBM, D / GBN), 256, 0, stream>>>(
        selHi, selLo, wkHi, wkLo, b_k, nullptr, kHiP, kLoP, B * TK, D, D);
    gemm_planes_kernel<<<dim3((B * TK) / GBM, D / GBN), 256, 0, stream>>>(
        selHi, selLo, wvHi, wvLo, b_v, nullptr, vHiP, vLoP, B * TK, D, D);

    sel_attn_kernel<<<dim3(S / 16, H, B), 32, 0, stream>>>(
        qHiP, qLoP, kHiP, kLoP, vHiP, vLoP, topidx, oHi, oLo, S);

    gemm_planes_kernel<<<dim3(Mq / GBM, D / GBN), 256, 0, stream>>>(
        oHi, oLo, woHi, woLo, b_o, out, nullptr, nullptr, Mq, D, D);
}
